// GAT_21835613733612
// MI455X (gfx1250) — matrix operand order repaired and checked by reading, not compiled, not run on hardware
//
#include <hip/hip_runtime.h>

// ---------------- problem constants ----------------
#define NNODE   100000
#define NEDGE   1000000
#define ETOT    (NEDGE + NNODE)       // with self loops
#define NFEAT   512
#define NHEAD   8
#define NHID    32
#define HC      (NHEAD * NHID)        // 256
#define NCLASS  40
#define SLOPE   0.2f

typedef __bf16 bf16_t;
typedef __attribute__((ext_vector_type(16))) bf16_t v16bf;
typedef __attribute__((ext_vector_type(8)))  float  v8f;

union AFrag { v16bf v; unsigned short u[16]; };
union BFrag { v16bf v; uint4 q[2]; };

__device__ __forceinline__ unsigned short f2bf(float f) {
    unsigned int u = __float_as_uint(f);
    unsigned int r = (u + 0x7FFFu + ((u >> 16) & 1u)) >> 16;   // RNE
    return (unsigned short)r;
}

// ---------------- CDNA5 async global->LDS copy (ASYNCcnt path) ----------------
#if __has_builtin(__builtin_amdgcn_global_load_async_to_lds_b128)
#define HAVE_ASYNC_LDS 1
#else
#define HAVE_ASYNC_LDS 0
#endif

typedef int v4i_t __attribute__((ext_vector_type(4)));
typedef __attribute__((address_space(3))) v4i_t lds_v4i;

__device__ __forceinline__ void async_cp16(const uint4* g, uint4* l) {
#if HAVE_ASYNC_LDS
    __builtin_amdgcn_global_load_async_to_lds_b128((v4i_t*)g, (lds_v4i*)l, 0, 0);
#else
    *l = *g;                                // sync fallback: short live range
#endif
}

__device__ __forceinline__ void wait_async0() {
#if HAVE_ASYNC_LDS
#if __has_builtin(__builtin_amdgcn_s_wait_asynccnt)
    __builtin_amdgcn_s_wait_asynccnt(0);
#else
    asm volatile("s_wait_asynccnt 0x0" ::: "memory");
#endif
#endif
}

// ---------------- utility kernels ----------------
__global__ void k_zero_i32(int* p, int n) {
    int t = blockIdx.x * blockDim.x + threadIdx.x;
    if (t < n) p[t] = 0;
}

// Pack W[K x Nw] fp32 row-major -> bf16 WMMA-B fragment order.
// idx = ((kk*NT + nt)*32 + lane)*16 + e ;  k = kk*32 + 16*(e>>3) + 8*(lane>>4) + (e&7) ; col = nt*16 + (lane&15)
__global__ void k_pack_b(const float* __restrict__ W, unsigned short* __restrict__ out,
                         int K, int Nw, int NT) {
    int t = blockIdx.x * blockDim.x + threadIdx.x;
    int total = (K / 32) * NT * 512;
    if (t >= total) return;
    int e    = t & 15;
    int lane = (t >> 4) & 31;
    int nt   = (t >> 9) % NT;
    int kk   = (t >> 9) / NT;
    int k    = kk * 32 + ((e >> 3) << 4) + ((lane >> 4) << 3) + (e & 7);
    int col  = nt * 16 + (lane & 15);
    float v = (col < Nw) ? W[(size_t)k * Nw + col] : 0.0f;
    out[t] = f2bf(v);
}

// ---------------- WMMA GEMM: async-LDS double-buffered B ----------------
// C[M x Nreal] = A[M x K](fp32, converted in-register) * Bpack(bf16 fragment order)
template <int NT>
__global__ __launch_bounds__(256)
void k_wmma_gemm(const float* __restrict__ A, const unsigned short* __restrict__ Bpack,
                 float* __restrict__ C, int M, int K, int Nreal, int ldc) {
    constexpr int FR  = NT * 64;                 // uint4 per stage
    constexpr int STG = (FR + 255) / 256;        // uint4 per thread per stage
    __shared__ uint4 sB[2 * FR];

    const int tid  = threadIdx.x;
    const int wave = tid >> 5;
    const int lane = tid & 31;
    const int half = lane >> 4;
    const int lr   = lane & 15;
    const int rowA = blockIdx.x * 128 + wave * 16 + lr;
    const bool arow = rowA < M;
    const int base0 = 8 * half;

    v8f acc[NT];
#pragma unroll
    for (int i = 0; i < NT; ++i) { v8f z = {0,0,0,0,0,0,0,0}; acc[i] = z; }

    const uint4* bg = (const uint4*)Bpack;
    const int ksteps = K >> 5;

    // stage helper: copy stage `kkS` of Bpack into LDS buffer `buf`
    auto stage = [&](int kkS, int buf) {
#pragma unroll
        for (int i = 0; i < STG; ++i) {
            int idx = tid + i * 256;
            if ((FR & 255) == 0 || idx < FR)
                async_cp16(bg + (size_t)kkS * FR + idx, &sB[buf * FR + idx]);
        }
    };

    // ---- prologue ----
    stage(0, 0);
    float4 pa[4] = {};
    if (arow) {
        const float4* ar = (const float4*)(A + (size_t)rowA * K + base0);
        pa[0] = ar[0]; pa[1] = ar[1]; pa[2] = ar[4]; pa[3] = ar[5];
    }
    wait_async0();
    __syncthreads();

    for (int kk = 0; kk < ksteps; ++kk) {
        // kick off next B stage (no VGPR footprint on the async path)
        if (kk + 1 < ksteps) stage(kk + 1, (kk + 1) & 1);

        // convert current A regs -> bf16 fragment (native v_cvt)
        AFrag af;
        if (arow) {
            af.v[0]=(bf16_t)pa[0].x; af.v[1]=(bf16_t)pa[0].y; af.v[2]=(bf16_t)pa[0].z; af.v[3]=(bf16_t)pa[0].w;
            af.v[4]=(bf16_t)pa[1].x; af.v[5]=(bf16_t)pa[1].y; af.v[6]=(bf16_t)pa[1].z; af.v[7]=(bf16_t)pa[1].w;
            af.v[8]=(bf16_t)pa[2].x; af.v[9]=(bf16_t)pa[2].y; af.v[10]=(bf16_t)pa[2].z; af.v[11]=(bf16_t)pa[2].w;
            af.v[12]=(bf16_t)pa[3].x; af.v[13]=(bf16_t)pa[3].y; af.v[14]=(bf16_t)pa[3].z; af.v[15]=(bf16_t)pa[3].w;
        } else {
#pragma unroll
            for (int e = 0; e < 16; ++e) af.v[e] = (bf16_t)0.0f;
        }
        // prefetch A regs for kk+1 (wait lands at next iteration's convert)
        if (arow && kk + 1 < ksteps) {
            const float4* ar = (const float4*)(A + (size_t)rowA * K + (kk + 1) * 32 + base0);
            pa[0] = ar[0]; pa[1] = ar[1]; pa[2] = ar[4]; pa[3] = ar[5];
        }

        // WMMA chain off LDS, ping-pong fragment buffers (no copy -> real pipelining)
        const uint4* sbuf = &sB[(kk & 1) * FR];
        BFrag b0, b1;
        b0.q[0] = sbuf[lane * 2];
        b0.q[1] = sbuf[lane * 2 + 1];
#pragma unroll
        for (int nt = 0; nt < NT; nt += 2) {
            if (nt + 1 < NT) {
                b1.q[0] = sbuf[(nt + 1) * 64 + lane * 2];
                b1.q[1] = sbuf[(nt + 1) * 64 + lane * 2 + 1];
            }
            acc[nt] = __builtin_amdgcn_wmma_f32_16x16x32_bf16(
                false, af.v, false, b0.v, (short)0, acc[nt], false, false);
            if (nt + 2 < NT) {
                b0.q[0] = sbuf[(nt + 2) * 64 + lane * 2];
                b0.q[1] = sbuf[(nt + 2) * 64 + lane * 2 + 1];
            }
            if (nt + 1 < NT)
                acc[nt + 1] = __builtin_amdgcn_wmma_f32_16x16x32_bf16(
                    false, af.v, false, b1.v, (short)0, acc[nt + 1], false, false);
        }

        if (kk + 1 < ksteps) wait_async0();   // next stage landed in LDS
        __syncthreads();
    }

#pragma unroll
    for (int nt = 0; nt < NT; ++nt) {
        const int col = nt * 16 + lr;
        if (col >= Nreal) continue;
#pragma unroll
        for (int g = 0; g < 8; ++g) {
            const int rowS = blockIdx.x * 128 + wave * 16 + (half ? 8 + g : g);
            if (rowS < M) C[(size_t)rowS * ldc + col] = acc[nt][g];
        }
    }
}

// ---------------- attention score kernels ----------------
__global__ void k_att1(const float* __restrict__ h, const float* __restrict__ asrc,
                       const float* __restrict__ adst,
                       float* __restrict__ esrc, float* __restrict__ edst) {
    int t = blockIdx.x * blockDim.x + threadIdx.x;        // t = n*8 + head
    if (t >= NNODE * NHEAD) return;
    int hd = t & 7;
    int n  = t >> 3;
    const float4* hp = (const float4*)(h + (size_t)n * HC + hd * NHID);
    const float4* sp = (const float4*)(asrc + hd * NHID);
    const float4* dp = (const float4*)(adst + hd * NHID);
    float es = 0.f, ed = 0.f;
#pragma unroll
    for (int i = 0; i < 8; ++i) {
        float4 v = hp[i], a = sp[i], b = dp[i];
        es += v.x * a.x + v.y * a.y + v.z * a.z + v.w * a.w;
        ed += v.x * b.x + v.y * b.y + v.z * b.z + v.w * b.w;
    }
    esrc[t] = es;
    edst[t] = ed;
}

__global__ void k_att2(const float* __restrict__ h, const float* __restrict__ asrc,
                       const float* __restrict__ adst,
                       float* __restrict__ esrc, float* __restrict__ edst) {
    int n = blockIdx.x * blockDim.x + threadIdx.x;
    if (n >= NNODE) return;
    float es = 0.f, ed = 0.f;
    const float* hp = h + (size_t)n * NCLASS;
    for (int j = 0; j < NCLASS; ++j) { float v = hp[j]; es += v * asrc[j]; ed += v * adst[j]; }
    esrc[n] = es;
    edst[n] = ed;
}

// ---------------- CSR build ----------------
__global__ void k_count(const int* __restrict__ ei, int* __restrict__ indeg) {
    int t = blockIdx.x * blockDim.x + threadIdx.x;
    if (t >= ETOT) return;
    int d = (t < NEDGE) ? ei[NEDGE + t] : (t - NEDGE);
    atomicAdd(&indeg[d], 1);
}

__global__ void k_scan(const int* __restrict__ indeg, int* __restrict__ offs,
                       int* __restrict__ cursor) {
    __shared__ int sdat[256];
    const int t = threadIdx.x;
    const int chunk = (NNODE + 255) / 256;
    const int lo = t * chunk, hi = min(NNODE, lo + chunk);
    int sum = 0;
    for (int i = lo; i < hi; ++i) sum += indeg[i];
    sdat[t] = sum;
    __syncthreads();
    for (int off = 1; off < 256; off <<= 1) {
        int v = (t >= off) ? sdat[t - off] : 0;
        __syncthreads();
        sdat[t] += v;
        __syncthreads();
    }
    int run = sdat[t] - sum;                 // exclusive base for this chunk
    for (int i = lo; i < hi; ++i) {
        offs[i] = run; cursor[i] = run;
        run += indeg[i];
    }
    if (t == 255) offs[NNODE] = sdat[255];
}

__global__ void k_fill(const int* __restrict__ ei, int* __restrict__ cursor,
                       int* __restrict__ csr_src) {
    int t = blockIdx.x * blockDim.x + threadIdx.x;
    if (t >= ETOT) return;
    int s, d;
    if (t < NEDGE) { s = ei[t]; d = ei[NEDGE + t]; } else { s = d = t - NEDGE; }
    int pos = atomicAdd(&cursor[d], 1);
    csr_src[pos] = s;
}

// ---------------- layer-1 aggregation: one wave per dst, online softmax ----------------
__global__ __launch_bounds__(256)
void k_agg1(const float* __restrict__ h, const float* __restrict__ esrc,
            const float* __restrict__ edst, const int* __restrict__ offs,
            const int* __restrict__ csr_src, const float* __restrict__ bias,
            float* __restrict__ out) {
    const int n = blockIdx.x * 8 + (threadIdx.x >> 5);
    if (n >= NNODE) return;
    const int lane = threadIdx.x & 31;
    const int hd   = lane >> 2;                 // 4 lanes per head
    const int colb = lane * 8;                  // 8 output cols per lane

    const float ed = edst[n * NHEAD + hd];
    float m = -__builtin_inff();
    float den = 0.f;
    float acc[8];
#pragma unroll
    for (int j = 0; j < 8; ++j) acc[j] = 0.f;

    const int lo = offs[n], hi = offs[n + 1];
    for (int i = lo; i < hi; ++i) {
        const int s = csr_src[i];
        float e = esrc[s * NHEAD + hd] + ed;
        e = (e > 0.f) ? e : SLOPE * e;
        if (e > m) {
            float f = __expf(m - e);
            den *= f;
#pragma unroll
            for (int j = 0; j < 8; ++j) acc[j] *= f;
            m = e;
        }
        const float w = __expf(e - m);
        den += w;
        const float4* hp = (const float4*)(h + (size_t)s * HC + colb);
        float4 a = hp[0], b = hp[1];
        acc[0] += w * a.x; acc[1] += w * a.y; acc[2] += w * a.z; acc[3] += w * a.w;
        acc[4] += w * b.x; acc[5] += w * b.y; acc[6] += w * b.z; acc[7] += w * b.w;
    }

    const float inv = 1.0f / den;
    float ov[8];
#pragma unroll
    for (int j = 0; j < 8; ++j) {
        float v = acc[j] * inv + bias[colb + j];
        ov[j] = (v > 0.f) ? v : (__expf(v) - 1.0f);      // ELU
    }
    float4 o0, o1;
    o0.x = ov[0]; o0.y = ov[1]; o0.z = ov[2]; o0.w = ov[3];
    o1.x = ov[4]; o1.y = ov[5]; o1.z = ov[6]; o1.w = ov[7];
    float4* op = (float4*)(out + (size_t)n * HC + colb);
    op[0] = o0; op[1] = o1;
}

// ---------------- layer-2 aggregation + log_softmax: one wave per dst ----------------
__global__ __launch_bounds__(256)
void k_agg2(const float* __restrict__ h, const float* __restrict__ esrc,
            const float* __restrict__ edst, const int* __restrict__ offs,
            const int* __restrict__ csr_src, const float* __restrict__ bias,
            float* __restrict__ logits) {
    const int n = blockIdx.x * 8 + (threadIdx.x >> 5);
    if (n >= NNODE) return;
    const int lane = threadIdx.x & 31;

    const float ed = edst[n];
    float m = -__builtin_inff();
    float den = 0.f, a0 = 0.f, a1 = 0.f;

    const int lo = offs[n], hi = offs[n + 1];
    for (int i = lo; i < hi; ++i) {
        const int s = csr_src[i];
        float e = esrc[s] + ed;
        e = (e > 0.f) ? e : SLOPE * e;
        if (e > m) {
            float f = __expf(m - e);
            den *= f; a0 *= f; a1 *= f;
            m = e;
        }
        const float w = __expf(e - m);
        den += w;
        const float* hp = h + (size_t)s * NCLASS;
        a0 += w * hp[lane];
        if (lane < 8) a1 += w * hp[lane + 32];
    }

    const float inv = 1.0f / den;
    float v0 = a0 * inv + bias[lane];
    float v1 = (lane < 8) ? (a1 * inv + bias[lane + 32]) : -__builtin_inff();

    // log_softmax across 40 classes held in (v0 lanes 0..31, v1 lanes 0..7)
    float rmax = fmaxf(v0, v1);
#pragma unroll
    for (int s = 16; s > 0; s >>= 1) rmax = fmaxf(rmax, __shfl_xor(rmax, s, 32));
    float ssum = __expf(v0 - rmax) + ((lane < 8) ? __expf(v1 - rmax) : 0.f);
#pragma unroll
    for (int s = 16; s > 0; s >>= 1) ssum += __shfl_xor(ssum, s, 32);
    const float lse = rmax + __logf(ssum);

    float* op = logits + (size_t)n * NCLASS;
    op[lane] = v0 - lse;
    if (lane < 8) op[lane + 32] = v1 - lse;
}

// ---------------- host launcher ----------------
static inline size_t alignup(size_t x) { return (x + 255) & ~(size_t)255; }

extern "C" void kernel_launch(void* const* d_in, const int* in_sizes, int n_in,
                              void* d_out, int out_size, void* d_ws, size_t ws_size,
                              hipStream_t stream) {
    const float* x      = (const float*)d_in[0];
    const int*   ei     = (const int*)d_in[1];
    const float* W1     = (const float*)d_in[2];
    const float* a_src1 = (const float*)d_in[3];
    const float* a_dst1 = (const float*)d_in[4];
    const float* b1     = (const float*)d_in[5];
    const float* W2     = (const float*)d_in[6];
    const float* a_src2 = (const float*)d_in[7];
    const float* a_dst2 = (const float*)d_in[8];
    const float* b2     = (const float*)d_in[9];

    float* logits = (float*)d_out;                          // [N, 40]
    float* h1     = (float*)d_out + (size_t)NNODE * NCLASS; // [N, 256] second output

    // workspace carve
    char* w = (char*)d_ws;
    size_t o = 0;
    float* h1pre  = (float*)(w + o); o = alignup(o + sizeof(float) * (size_t)NNODE * HC);
    float* h2pre  = (float*)(w + o); o = alignup(o + sizeof(float) * (size_t)NNODE * NCLASS);
    float* esrc1  = (float*)(w + o); o = alignup(o + sizeof(float) * (size_t)NNODE * NHEAD);
    float* edst1  = (float*)(w + o); o = alignup(o + sizeof(float) * (size_t)NNODE * NHEAD);
    float* esrc2  = (float*)(w + o); o = alignup(o + sizeof(float) * (size_t)NNODE);
    float* edst2  = (float*)(w + o); o = alignup(o + sizeof(float) * (size_t)NNODE);
    int*   indeg  = (int*)(w + o);   o = alignup(o + sizeof(int) * (size_t)NNODE);
    int*   offs   = (int*)(w + o);   o = alignup(o + sizeof(int) * ((size_t)NNODE + 1));
    int*   cursor = (int*)(w + o);   o = alignup(o + sizeof(int) * (size_t)NNODE);
    int*   csrs   = (int*)(w + o);   o = alignup(o + sizeof(int) * (size_t)ETOT);
    unsigned short* bp1 = (unsigned short*)(w + o); o = alignup(o + sizeof(short) * (size_t)(NFEAT/32) * 16 * 512);
    unsigned short* bp2 = (unsigned short*)(w + o); o = alignup(o + sizeof(short) * (size_t)(HC/32)   * 3  * 512);

    const int T = 256;
    // CSR build
    k_zero_i32<<<(NNODE + T - 1) / T, T, 0, stream>>>(indeg, NNODE);
    k_count<<<(ETOT + T - 1) / T, T, 0, stream>>>(ei, indeg);
    k_scan<<<1, 256, 0, stream>>>(indeg, offs, cursor);
    k_fill<<<(ETOT + T - 1) / T, T, 0, stream>>>(ei, cursor, csrs);

    // pack weights to bf16 WMMA-B fragment layout
    {
        int tot1 = (NFEAT / 32) * 16 * 512;
        k_pack_b<<<(tot1 + T - 1) / T, T, 0, stream>>>(W1, bp1, NFEAT, HC, 16);
        int tot2 = (HC / 32) * 3 * 512;
        k_pack_b<<<(tot2 + T - 1) / T, T, 0, stream>>>(W2, bp2, HC, NCLASS, 3);
    }

    const int gblk = (NNODE + 127) / 128;
    // layer 1
    k_wmma_gemm<16><<<gblk, 256, 0, stream>>>(x, bp1, h1pre, NNODE, NFEAT, HC, HC);
    k_att1<<<(NNODE * NHEAD + T - 1) / T, T, 0, stream>>>(h1pre, a_src1, a_dst1, esrc1, edst1);
    k_agg1<<<(NNODE + 7) / 8, 256, 0, stream>>>(h1pre, esrc1, edst1, offs, csrs, b1, h1);
    // layer 2 (reads h1 from d_out region)
    k_wmma_gemm<3><<<gblk, 256, 0, stream>>>(h1, bp2, h2pre, NNODE, HC, NCLASS, NCLASS);
    k_att2<<<(NNODE + T - 1) / T, T, 0, stream>>>(h2pre, a_src2, a_dst2, esrc2, edst2);
    k_agg2<<<(NNODE + 7) / 8, 256, 0, stream>>>(h2pre, esrc2, edst2, offs, csrs, b2, logits);

    (void)in_sizes; (void)n_in; (void)out_size; (void)ws_size;
}
